// MultiHeadAttention_17763984736858
// MI455X (gfx1250) — matrix operand order repaired and checked by reading, not compiled, not run on hardware
//
#include <hip/hip_runtime.h>
#include <hip/hip_bf16.h>

typedef __attribute__((ext_vector_type(16))) _Float16 v16h;
typedef __attribute__((ext_vector_type(8)))  _Float16 v8h;
typedef __attribute__((ext_vector_type(8)))  float    v8f;

#define D_MODEL  1024
#define NHEAD    16
#define HEAD_DIM 64
#define SEQ      2048
#define NBATCH   2
#define NTOK     (NBATCH * SEQ)     // 4096
#define QKV_N    (3 * D_MODEL)      // 3072
#define WIN      256

#define AS1 __attribute__((address_space(1)))
#define AS3 __attribute__((address_space(3)))

#if defined(__has_builtin)
#  if __has_builtin(__builtin_amdgcn_global_load_async_to_lds_b128)
#    define HAVE_ASYNC 1
#  endif
#endif
#ifndef HAVE_ASYNC
#  define HAVE_ASYNC 0
#endif

// ---------------------------------------------------------------------------
// helpers
// ---------------------------------------------------------------------------
static __device__ __forceinline__ v16h load_frag16(const _Float16* p0, const _Float16* p1) {
  v16h a;
  *(v8h*)&a       = *(const v8h*)p0;   // 16B
  *((v8h*)&a + 1) = *(const v8h*)p1;   // 16B
  return a;
}

static __device__ __forceinline__ v8f wmma_f16(v16h a, v16h b, v8f c) {
  return __builtin_amdgcn_wmma_f32_16x16x32_f16(false, a, false, b, (short)0, c, false, false);
}

// exact param type of the async builtins: GCC-style int vector, AS1/AS3 pointers
typedef int v4i_g __attribute__((vector_size(16)));

static __device__ __forceinline__ void async_b128(const _Float16* g, _Float16* l) {
#if HAVE_ASYNC
  // generic->AS1: same 64-bit value; generic->AS3: LDS offset is low 32 bits (ISA 10.2)
  __builtin_amdgcn_global_load_async_to_lds_b128(
      (AS1 v4i_g*)(uintptr_t)g,
      (AS3 v4i_g*)(uint32_t)(uintptr_t)l,
      0, 0);
#else
  *(v8h*)l = *(const v8h*)g;
#endif
}

static __device__ __forceinline__ void wait_async() {
#if HAVE_ASYNC
#  if __has_builtin(__builtin_amdgcn_s_wait_asynccnt)
  __builtin_amdgcn_s_wait_asynccnt(0);
#  else
  asm volatile("s_wait_asynccnt 0x0" ::: "memory");
#  endif
#endif
}

// ---------------------------------------------------------------------------
// f32 -> f16 conversion
// ---------------------------------------------------------------------------
__global__ void cvt_f32_f16(const float* __restrict__ in, _Float16* __restrict__ out, int n) {
  int i = blockIdx.x * blockDim.x + threadIdx.x;
  if (i < n) out[i] = (_Float16)in[i];
}

// ---------------------------------------------------------------------------
// C[m][n] = sum_k A[m][k] * B[n][k]   (C = A * B^T)
// A: MxK f16 row-major, B: NxK f16 row-major.
// Block: 256 thr = 8 waves (4M x 2N). Block tile 128x128, per-wave 32x64.
// K-slices of A and B (128 rows x 32 halves, rows padded to 40 halves = 80B,
// 20-bank stride -> conflict-free) staged in LDS via async-to-LDS engine,
// double-buffered; 8 WMMAs per wave per k-step from ds_load_b128.
// ---------------------------------------------------------------------------
template <bool BIAS>
__global__ void __launch_bounds__(256, 1)
gemm_wmma_lds(const _Float16* __restrict__ A, const _Float16* __restrict__ B,
              const float* __restrict__ bias, void* __restrict__ Cout,
              int K, int N) {
  __shared__ __align__(16) _Float16 Ash[2][128][40];
  __shared__ __align__(16) _Float16 Bsh[2][128][40];

  const int tid  = threadIdx.x;
  const int lane = tid & 31, wid = tid >> 5;
  const int wm = wid & 3, wn = wid >> 2;            // 4 M-waves x 2 N-waves
  const int mblk = blockIdx.y * 128, nblk = blockIdx.x * 128;
  const int col = lane & 15, g = lane >> 4;

  // staging map: thread t -> row sr = t/2, 32B chunk (t&1): halves [sc, sc+16)
  const int sr = tid >> 1, sc = (tid & 1) * 16;
  const _Float16* gA = A + (size_t)(mblk + sr) * K + sc;
  const _Float16* gB = B + (size_t)(nblk + sr) * K + sc;

  v8f acc[2][4] = {};

  auto stage = [&](int buf, int k0) {
    async_b128(gA + k0,     &Ash[buf][sr][sc]);
    async_b128(gA + k0 + 8, &Ash[buf][sr][sc + 8]);
    async_b128(gB + k0,     &Bsh[buf][sr][sc]);
    async_b128(gB + k0 + 8, &Bsh[buf][sr][sc + 8]);
  };

  stage(0, 0);
  const int nk = K >> 5;
#pragma unroll 1
  for (int i = 0; i < nk; ++i) {
    const int buf = i & 1;
    wait_async();            // this wave's async writes for tile i are in LDS
    __syncthreads();         // make them visible block-wide
    if (i + 1 < nk) stage(1 - buf, (i + 1) * 32);

    // A-frags: lane l holds row (l&15); halves 0..7 = K g*8.., 8..15 = K 16+g*8..
    v16h af[2];
#pragma unroll
    for (int mt = 0; mt < 2; ++mt) {
      const _Float16* ar = &Ash[buf][wm * 32 + mt * 16 + col][0];
      af[mt] = load_frag16(ar + g * 8, ar + 16 + g * 8);
    }
    // B-frags: lane n holds halves 0..7 = K g*8.., 8..15 = K 16+g*8..
#pragma unroll
    for (int nt = 0; nt < 4; ++nt) {
      const _Float16* br = &Bsh[buf][wn * 64 + nt * 16 + col][0];
      v16h b = load_frag16(br + g * 8, br + 16 + g * 8);
      acc[0][nt] = wmma_f16(af[0], b, acc[0][nt]);
      acc[1][nt] = wmma_f16(af[1], b, acc[1][nt]);
    }

    __syncthreads();         // all waves done reading buf before it is re-staged
  }

  const int rb = g * 8;      // C layout: VGPR r, lanes 0-15: M=r; lanes 16-31: M=r+8
#pragma unroll
  for (int mt = 0; mt < 2; ++mt) {
    const int mbase = mblk + wm * 32 + mt * 16 + rb;
#pragma unroll
    for (int nt = 0; nt < 4; ++nt) {
      const int n = nblk + wn * 64 + nt * 16 + col;
      if constexpr (BIAS) {
        float bv = bias[n];
        float* C = (float*)Cout;
#pragma unroll
        for (int r = 0; r < 8; ++r)
          C[(size_t)(mbase + r) * N + n] = acc[mt][nt][r] + bv;
      } else {
        _Float16* C = (_Float16*)Cout;
#pragma unroll
        for (int r = 0; r < 8; ++r)
          C[(size_t)(mbase + r) * N + n] = (_Float16)acc[mt][nt][r];
      }
    }
  }
}

// ---------------------------------------------------------------------------
// RoPE on q,k (in-place in qkv f16 buffer) + build V^T [nb][h][d][t] (f16).
// One thread per (token, head, d); d<32 threads rotate the (d, d+32) pair.
// ---------------------------------------------------------------------------
__global__ void rope_and_vt(_Float16* __restrict__ qkv, _Float16* __restrict__ vT) {
  int idx = blockIdx.x * blockDim.x + threadIdx.x;   // 4096*16*64 threads
  int d   = idx & 63;
  int h   = (idx >> 6) & 15;
  int tok = idx >> 10;            // 0..4095
  int nb  = tok >> 11;
  int t   = tok & (SEQ - 1);

  _Float16* base = qkv + (size_t)tok * QKV_N + h * HEAD_DIM;

  // V transpose: vT[((nb*16+h)*64+d)][t]
  vT[(((size_t)(nb * NHEAD + h) * HEAD_DIM + d) << 11) + t] = base[2 * D_MODEL + d];

  if (d < 32) {
    // inv_freq = 10000^(-d/32) = exp(-d * ln(10000)/32)
    float inv = __expf(-(float)d * 0.28782313662425575f);
    float ang = (float)t * inv;
    float c = cosf(ang), s = sinf(ang);

    float q1 = (float)base[d], q2 = (float)base[d + 32];
    base[d]      = (_Float16)(q1 * c - q2 * s);
    base[d + 32] = (_Float16)(q1 * s + q2 * c);

    float k1 = (float)base[D_MODEL + d], k2 = (float)base[D_MODEL + d + 32];
    base[D_MODEL + d]      = (_Float16)(k1 * c - k2 * s);
    base[D_MODEL + d + 32] = (_Float16)(k1 * s + k2 * c);
  }
}

// ---------------------------------------------------------------------------
// Sliding-window flash attention. One wave per 16-query tile of one (nb,h).
// Block = 128 thr = 4 waves. Online softmax; P re-layout C->A via LDS.
// ---------------------------------------------------------------------------
#define AW 4
__global__ void __launch_bounds__(32 * AW, 1)
attn_wmma(const _Float16* __restrict__ qkv, const _Float16* __restrict__ vT,
          _Float16* __restrict__ attnh) {
  __shared__ __align__(16) _Float16 lp[AW][16][40];   // 16 x 32 P tile, padded rows (80B stride)

  const int lane = threadIdx.x & 31, wid = threadIdx.x >> 5;
  const int tile = blockIdx.x * AW + wid;             // 0..4095
  const int it = tile & 127;
  const int h  = (tile >> 7) & 15;
  const int nb = tile >> 11;
  const int i0 = it * 16;
  const int col = lane & 15, g = lane >> 4, rb = g * 8;

  const _Float16* qbase = qkv + (size_t)(nb * SEQ) * QKV_N + h * HEAD_DIM;           // q
  const _Float16* kbase = qbase + D_MODEL;                                           // k
  const _Float16* vbase = vT + ((size_t)(nb * NHEAD + h) * HEAD_DIM << 11);          // v^T rows

  // Q A-fragments (row = i0+col per A layout), K-dim = head dim, 2 chunks of 32
  const _Float16* qrow = qbase + (size_t)(i0 + col) * QKV_N;
  v16h qa0 = load_frag16(qrow + g * 16,      qrow + g * 16 + 8);
  v16h qa1 = load_frag16(qrow + 32 + g * 16, qrow + 40 + g * 16);

  v8f o[4] = {};
  float mi[8], li[8];
#pragma unroll
  for (int r = 0; r < 8; ++r) { mi[r] = -__builtin_inff(); li[r] = 0.f; }

  const float scale = 0.125f;   // 1/sqrt(64)
  int jlo = i0 - WIN; if (jlo < 0) jlo = 0;
  jlo &= ~31;

#pragma unroll 1
  for (int jt = jlo; jt <= i0 + 15; jt += 32) {
    // ---- S = Q K^T for key tiles [jt, jt+16) and [jt+16, jt+32) ----
    v8f s0a = {}, s1a = {};
    {
      int tk0 = jt + col;      if (tk0 > SEQ - 1) tk0 = SEQ - 1;
      const _Float16* kr0 = kbase + (size_t)tk0 * QKV_N;
      v16h b = load_frag16(kr0 + g * 16, kr0 + g * 16 + 8);
      s0a = wmma_f16(qa0, b, s0a);
      b = load_frag16(kr0 + 32 + g * 16, kr0 + 32 + g * 16 + 8);
      s0a = wmma_f16(qa1, b, s0a);

      int tk1 = jt + 16 + col; if (tk1 > SEQ - 1) tk1 = SEQ - 1;
      const _Float16* kr1 = kbase + (size_t)tk1 * QKV_N;
      b = load_frag16(kr1 + g * 16, kr1 + g * 16 + 8);
      s1a = wmma_f16(qa0, b, s1a);
      b = load_frag16(kr1 + 32 + g * 16, kr1 + 32 + g * 16 + 8);
      s1a = wmma_f16(qa1, b, s1a);
    }

    // ---- mask + online softmax; O rescale folded in (row m = rb+r) ----
#pragma unroll
    for (int r = 0; r < 8; ++r) {
      int   i  = i0 + rb + r;
      int   j0 = jt + col, j1 = jt + 16 + col;
      float s0 = (j0 <= i && j0 >= i - WIN) ? s0a[r] * scale : -__builtin_inff();
      float s1 = (j1 <= i && j1 >= i - WIN) ? s1a[r] * scale : -__builtin_inff();

      float rm = fmaxf(s0, s1);
#pragma unroll
      for (int msk = 1; msk < 16; msk <<= 1) rm = fmaxf(rm, __shfl_xor(rm, msk, 32));

      float mn = fmaxf(mi[r], rm);
      bool  dead = (mn == -__builtin_inff());
      float c  = dead ? 1.0f : __expf(mi[r] - mn);
      float e0 = dead ? 0.0f : __expf(s0 - mn);
      float e1 = dead ? 0.0f : __expf(s1 - mn);

      float rs = e0 + e1;
#pragma unroll
      for (int msk = 1; msk < 16; msk <<= 1) rs += __shfl_xor(rs, msk, 32);

      li[r] = li[r] * c + rs;
      mi[r] = mn;
      o[0][r] *= c; o[1][r] *= c; o[2][r] *= c; o[3][r] *= c;

      lp[wid][rb + r][col]      = (_Float16)e0;
      lp[wid][rb + r][col + 16] = (_Float16)e1;
    }

    // wave-internal LDS visibility: stores (C layout) -> loads (A layout)
    asm volatile("s_wait_dscnt 0" ::: "memory");

    // P A-fragment: row = col, halves 0..15 = K g*16..g*16+15
    v16h pa = load_frag16(&lp[wid][col][g * 16], &lp[wid][col][g * 16 + 8]);

    // ---- O += P @ V ----
    int jc = jt + g * 16; if (jc > SEQ - 16) jc = SEQ - 16;
#pragma unroll
    for (int dt = 0; dt < 4; ++dt) {
      const _Float16* vrow = vbase + ((size_t)(dt * 16 + col) << 11) + jc;
      v16h vb = load_frag16(vrow, vrow + 8);
      o[dt] = wmma_f16(pa, vb, o[dt]);
    }
  }

  // ---- finalize: O /= l, store f16 row-major [token][h*64 + d] ----
  _Float16* obase = attnh + (size_t)(nb * SEQ) * D_MODEL + h * HEAD_DIM;
#pragma unroll
  for (int r = 0; r < 8; ++r) {
    float inv = 1.0f / li[r];
#pragma unroll
    for (int dt = 0; dt < 4; ++dt)
      obase[(size_t)(i0 + rb + r) * D_MODEL + dt * 16 + col] = (_Float16)(o[dt][r] * inv);
  }
}

// ---------------------------------------------------------------------------
// launch
// ---------------------------------------------------------------------------
extern "C" void kernel_launch(void* const* d_in, const int* in_sizes, int n_in,
                              void* d_out, int out_size, void* d_ws, size_t ws_size,
                              hipStream_t stream) {
  const float* x     = (const float*)d_in[0];   // [2,2048,1024]
  const float* Wqkv  = (const float*)d_in[1];   // [3072,1024]
  const float* out_w = (const float*)d_in[2];   // [1024,1024]
  const float* out_b = (const float*)d_in[3];   // [1024]
  float* out = (float*)d_out;                   // [2,2048,1024]

  char* ws = (char*)d_ws;
  size_t off = 0;
  auto carve = [&](size_t elems) {
    _Float16* p = (_Float16*)(ws + off);
    off += ((elems * sizeof(_Float16) + 255) & ~(size_t)255);
    return p;
  };
  _Float16* xh   = carve((size_t)NTOK * D_MODEL);                       // 8 MB
  _Float16* wh   = carve((size_t)QKV_N * D_MODEL);                      // 6 MB
  _Float16* owh  = carve((size_t)D_MODEL * D_MODEL);                    // 2 MB
  _Float16* qkvh = carve((size_t)NTOK * QKV_N);                         // 24 MB
  _Float16* vTh  = carve((size_t)NBATCH * NHEAD * HEAD_DIM * SEQ);      // 8 MB
  _Float16* atth = carve((size_t)NTOK * D_MODEL);                       // 8 MB

  int n1 = NTOK * D_MODEL;
  cvt_f32_f16<<<(n1 + 255) / 256, 256, 0, stream>>>(x, xh, n1);
  int n2 = QKV_N * D_MODEL;
  cvt_f32_f16<<<(n2 + 255) / 256, 256, 0, stream>>>(Wqkv, wh, n2);
  int n3 = D_MODEL * D_MODEL;
  cvt_f32_f16<<<(n3 + 255) / 256, 256, 0, stream>>>(out_w, owh, n3);

  // QKV projection: [4096,1024] x [3072,1024]^T -> [4096,3072] f16
  gemm_wmma_lds<false><<<dim3(QKV_N / 128, NTOK / 128), 256, 0, stream>>>(
      xh, wh, nullptr, qkvh, D_MODEL, QKV_N);

  // RoPE q,k in-place + build V^T
  rope_and_vt<<<(NTOK * NHEAD * HEAD_DIM) / 256, 256, 0, stream>>>(qkvh, vTh);

  // Sliding-window attention -> [4096,1024] f16
  attn_wmma<<<(NBATCH * NHEAD * (SEQ / 16)) / AW, 32 * AW, 0, stream>>>(qkvh, vTh, atth);

  // Output projection + bias -> d_out f32
  gemm_wmma_lds<true><<<dim3(D_MODEL / 128, NTOK / 128), 256, 0, stream>>>(
      atth, owh, out_b, out, D_MODEL, D_MODEL);
}
